// Codebook_10325101380280
// MI455X (gfx1250) — matrix operand order repaired and checked by reading, not compiled, not run on hardware
//
#include <hip/hip_runtime.h>

// ---------------------------------------------------------------------------
// VQ-VAE codebook quantization for MI455X (gfx1250, wave32, WMMA).
//   z   : [64,256,32,32] f32   (N = 65536 latent vectors, D = 256)
//   emb : [2048,256]     f32   (K = 2048 codes)
// Distance GEMM runs on v_wmma_f32_16x16x32_f16 using a double-f16 split
// (zh+zl)·(eh+el) ≈ zh·eh + zl·eh + zh·el  -> argmin matches f32 reference.
// emb tiles stream into LDS via GLOBAL_LOAD_ASYNC_TO_LDS_B128 (ASYNCcnt),
// double-buffered so the fill of stage n+1 overlaps the WMMAs of stage n.
// ---------------------------------------------------------------------------

typedef __attribute__((ext_vector_type(16))) _Float16 v16h;
typedef __attribute__((ext_vector_type(8)))  float    v8f;
typedef __attribute__((ext_vector_type(4)))  unsigned uv4;
typedef __attribute__((ext_vector_type(4)))  float    fv4;

#define NB      64
#define DIM     256
#define PP      1024            // 32*32
#define NVEC    65536           // 64*1024
#define KCB     2048
#define ZQ_ELE  16777216        // 64*256*1024

#define IDX1_OFF 16777216
#define LOSS_OFF 16842752
#define IDX2_OFF 16842753

// workspace layout (bytes)
#define WS_EMB16   0                      // 2048*512*2  = 2 MB   (eh || el)
#define WS_ZF16    2097152                // 65536*512*2 = 64 MB  (zh || zl)
#define WS_ENORM   69206016               // 2048*4
#define WS_IDX     69214208               // 65536*4
#define WS_PART    69476352               // 1024*4

#define BROW    1040                      // 1024B emb16 row + 16B pad (bank-tilt)

union H2  { _Float16 h[2]; unsigned u; };
union H8  { _Float16 h[8]; uv4 q; };
union AV  { v16h v; uv4 q[2]; };

__device__ __forceinline__ void wait_async0() {
#if __has_builtin(__builtin_amdgcn_s_wait_asynccnt)
  __builtin_amdgcn_s_wait_asynccnt(0);
#else
  asm volatile("s_wait_asynccnt 0x0" ::: "memory");
#endif
}

// Issue one 16-row (16KB) emb16 stage as 4 async b128 copies per thread.
// VDST operand = wave-relative LDS byte offset; VADDR = 64-bit global address.
__device__ __forceinline__ void fill_stage_async(const unsigned char* emb16b,
                                                 unsigned ldsbase, int n0, int t) {
#pragma unroll
  for (int q = 0; q < 4; ++q) {
    int cid = q * 256 + t;                       // 0..1023 (16B chunks)
    int row = cid >> 6;                          // 0..15
    unsigned lds_off = ldsbase + (unsigned)(cid * 16 + row * 16);  // +16B pad/row
    const unsigned char* g = emb16b + (size_t)n0 * 1024 + (size_t)cid * 16;
    asm volatile("global_load_async_to_lds_b128 %0, %1, off"
                 :: "v"(lds_off), "v"(g)
                 : "memory");
  }
}

// ---------------------------------------------------------------------------
// K1: emb f32 -> (eh, el) f16 split + enorm = ||e||^2 (exact f32).
// ---------------------------------------------------------------------------
__global__ __launch_bounds__(256) void k_prep_emb(const float* __restrict__ emb,
                                                  _Float16* __restrict__ emb16,
                                                  float* __restrict__ enorm) {
  const int row  = blockIdx.x * 8 + (threadIdx.x >> 5);
  const int lane = threadIdx.x & 31;
  const float* src = emb + (size_t)row * DIM + lane * 8;
  fv4 v0 = *(const fv4*)(src);
  fv4 v1 = *(const fv4*)(src + 4);
  H8 hi_, lo_;
  float ss = 0.f;
#pragma unroll
  for (int i = 0; i < 8; ++i) {
    float v = (i < 4) ? v0[i] : v1[i - 4];
    ss += v * v;
    _Float16 h = (_Float16)v;
    hi_.h[i] = h;
    lo_.h[i] = (_Float16)(v - (float)h);
  }
  _Float16* dst = emb16 + (size_t)row * 512 + lane * 8;
  *(uv4*)(dst)       = hi_.q;          // eh at cols [0,256)
  *(uv4*)(dst + 256) = lo_.q;          // el at cols [256,512)
#pragma unroll
  for (int off = 16; off >= 1; off >>= 1) ss += __shfl_xor(ss, off, 32);
  if (lane == 0) enorm[row] = ss;
}

// ---------------------------------------------------------------------------
// K2: z [b,c,hw] f32 -> zf16 [n=b*1024+hw][512] = (zh || zl).
// ---------------------------------------------------------------------------
__global__ __launch_bounds__(256) void k_transpose_z(const float* __restrict__ z,
                                                     _Float16* __restrict__ zf16) {
  __shared__ float tile[32][33];
  const int b = blockIdx.z, ct = blockIdx.y, pt = blockIdx.x;
  const int t = threadIdx.x;
  const float* src = z + ((size_t)(b * DIM + ct * 32)) * PP + pt * 32;
  const int tp = t & 31;
#pragma unroll
  for (int i = 0; i < 4; ++i) {
    int c = (t >> 5) + i * 8;
    tile[c][tp] = src[(size_t)c * PP + tp];
  }
  __syncthreads();
  const int cpair = (t & 15) * 2;
#pragma unroll
  for (int i = 0; i < 2; ++i) {
    int p = (t >> 4) + i * 16;
    float v0 = tile[cpair][p], v1 = tile[cpair + 1][p];
    H2 h, l;
    h.h[0] = (_Float16)v0;             h.h[1] = (_Float16)v1;
    l.h[0] = (_Float16)(v0 - (float)h.h[0]);
    l.h[1] = (_Float16)(v1 - (float)h.h[1]);
    size_t rowbase = ((size_t)(b * PP + pt * 32 + p)) * 512;
    *(unsigned*)(zf16 + rowbase +       ct * 32 + cpair) = h.u;   // zh
    *(unsigned*)(zf16 + rowbase + 256 + ct * 32 + cpair) = l.u;   // zl
  }
}

// ---------------------------------------------------------------------------
// K3: WMMA distance GEMM + per-row argmin.
// Block = 8 waves; wave w owns an M-tile of 16 z rows (A resident: 16 v16h).
// Per 16-code N-tile: 24 WMMAs (zh*eh + zl*eh + zh*el), fp32 accum,
// score = ||e||^2 - 2*dot. emb16 tiles double-buffer through LDS via async.
// ---------------------------------------------------------------------------
__global__ __launch_bounds__(256, 1) void k_argmin(const _Float16* __restrict__ zf16,
                                                   const _Float16* __restrict__ emb16,
                                                   const float* __restrict__ enorm,
                                                   unsigned* __restrict__ idx_ws,
                                                   float* __restrict__ out_idx1,
                                                   float* __restrict__ out_idx2) {
  __shared__ __align__(16) unsigned char ldsB[2][16 * BROW];   // 2 x 16.25KB

  const int t = threadIdx.x, lane = t & 31, wave = t >> 5;
  const int lhalf = lane & 15, hi = lane >> 4;
  const int mbase = (blockIdx.x * 8 + wave) * 16;

  // A fragments (16-bit A 16x32 layout): lanes 0-15 row=lane, K {0-7,16-23};
  // lanes 16-31 row=lane-16, K {8-15,24-31}. zf16 row-major makes this 2x b128.
  AV a[16];
  const _Float16* zr = zf16 + (size_t)(mbase + lhalf) * 512;
#pragma unroll
  for (int i = 0; i < 16; ++i) {
    int col = i * 32 + hi * 8;
    a[i].q[0] = *(const uv4*)(zr + col);
    a[i].q[1] = *(const uv4*)(zr + col + 16);
  }

  float    minv[8];
  unsigned mini[8];
#pragma unroll
  for (int r = 0; r < 8; ++r) { minv[r] = 3.4e38f; mini[r] = 0u; }

  const unsigned char* emb16b = (const unsigned char*)emb16;
  const unsigned lds0 = (unsigned)(uintptr_t)&ldsB[0][0];
  const unsigned lds1 = (unsigned)(uintptr_t)&ldsB[1][0];

  fill_stage_async(emb16b, lds0, 0, t);               // prologue: stage 0 -> buf 0

  for (int nt = 0; nt < 128; ++nt) {
    wait_async0();                                    // own async stores to buf[nt&1] done
    __syncthreads();                                  // all waves' fills done + prev reads done
    if (nt + 1 < 128)                                 // overlap: fill buf[(nt+1)&1] with stage nt+1
      fill_stage_async(emb16b, (nt & 1) ? lds0 : lds1, (nt + 1) * 16, t);

    const unsigned char* brow = &ldsB[nt & 1][0] + lhalf * BROW + hi * 16;
    v8f acc = {};
    // pass 1+2: eh chunks (B frag reused): zh*eh and zl*eh
#pragma unroll
    for (int kc = 0; kc < 8; ++kc) {
      AV bf;
      bf.q[0] = *(const uv4*)(brow + kc * 64);
      bf.q[1] = *(const uv4*)(brow + kc * 64 + 32);
      acc = __builtin_amdgcn_wmma_f32_16x16x32_f16(false, a[kc].v,     false, bf.v,
                                                   (short)0, acc, false, false);
      acc = __builtin_amdgcn_wmma_f32_16x16x32_f16(false, a[kc + 8].v, false, bf.v,
                                                   (short)0, acc, false, false);
    }
    // pass 3: el chunks: zh*el
#pragma unroll
    for (int kc = 0; kc < 8; ++kc) {
      AV bf;
      bf.q[0] = *(const uv4*)(brow + (kc + 8) * 64);
      bf.q[1] = *(const uv4*)(brow + (kc + 8) * 64 + 32);
      acc = __builtin_amdgcn_wmma_f32_16x16x32_f16(false, a[kc].v, false, bf.v,
                                                   (short)0, acc, false, false);
    }
    const float    en   = enorm[nt * 16 + lhalf];     // L2-resident, 64B/wave
    const unsigned cand = (unsigned)(nt * 16 + lhalf);
#pragma unroll
    for (int r = 0; r < 8; ++r) {
      float s = en - 2.0f * acc[r];                   // D layout: VGPR r -> M = r + 8*hi
      if (s < minv[r]) { minv[r] = s; mini[r] = cand; }
    }
  }

  // reduce across the 16 lanes of each half (they share the same 8 M rows);
  // tie-break on smaller index to match jnp.argmin (first minimum).
#pragma unroll
  for (int off = 1; off < 16; off <<= 1) {
#pragma unroll
    for (int r = 0; r < 8; ++r) {
      float    ov = __shfl_xor(minv[r], off, 32);
      unsigned oi = (unsigned)__shfl_xor((int)mini[r], off, 32);
      if (ov < minv[r] || (ov == minv[r] && oi < mini[r])) { minv[r] = ov; mini[r] = oi; }
    }
  }
  if (lhalf == 0) {
    int base = mbase + hi * 8;               // lane 0 -> rows 0..7, lane 16 -> 8..15
#pragma unroll
    for (int r = 0; r < 8; ++r) {
      unsigned id = mini[r];
      idx_ws[base + r]   = id;
      out_idx1[base + r] = (float)id;
      out_idx2[base + r] = (float)id;
    }
  }
}

// ---------------------------------------------------------------------------
// K4: gather z_q = emb[idx] (exact f32), straight-through output written in
// the reference's exact op order zp + (z_q - zp), per-block loss partials.
// ---------------------------------------------------------------------------
__global__ __launch_bounds__(256) void k_gather(const float* __restrict__ z,
                                                const float* __restrict__ emb,
                                                const unsigned* __restrict__ idx_ws,
                                                float* __restrict__ out_zq,
                                                float* __restrict__ partials) {
  __shared__ float red[256];
  const int blk = blockIdx.x;                // 1024 blocks
  const int b = blk >> 4, pc = blk & 15;
  const int t = threadIdx.x;
  const int pl = t & 63, cq = t >> 6;
  const unsigned id = idx_ws[b * PP + pc * 64 + pl];
  const float* erow = emb + (size_t)id * DIM;
  const size_t pbase = (size_t)b * (DIM * PP) + pc * 64 + pl;
  float accs = 0.f;
#pragma unroll 4
  for (int cg = 0; cg < 64; ++cg) {
    int c = cq * 64 + cg;
    size_t o = pbase + (size_t)c * PP;
    float zq = erow[c];
    float zv = z[o];
    float d = zq - zv;
    out_zq[o] = zv + d;                      // zp + sg(z_q - zp)
    accs += d * d;
  }
  red[t] = accs;
  __syncthreads();
  for (int s = 128; s > 0; s >>= 1) { if (t < s) red[t] += red[t + s]; __syncthreads(); }
  if (t == 0) partials[blk] = red[0];
}

// K5: deterministic final loss reduction: (1+BETA) * sum/N_total.
__global__ __launch_bounds__(256) void k_loss(const float* __restrict__ partials,
                                              float* __restrict__ out_loss) {
  __shared__ float red[256];
  const int t = threadIdx.x;
  float s = 0.f;
  for (int i = t; i < 1024; i += 256) s += partials[i];
  red[t] = s;
  __syncthreads();
  for (int st = 128; st > 0; st >>= 1) { if (t < st) red[t] += red[t + st]; __syncthreads(); }
  if (t == 0) out_loss[0] = red[0] * (1.25f / 16777216.0f);
}

// ---------------------------------------------------------------------------
extern "C" void kernel_launch(void* const* d_in, const int* in_sizes, int n_in,
                              void* d_out, int out_size, void* d_ws, size_t ws_size,
                              hipStream_t stream) {
  const float* z   = (const float*)d_in[0];
  const float* emb = (const float*)d_in[1];
  float* out = (float*)d_out;

  unsigned char* ws = (unsigned char*)d_ws;
  _Float16* emb16 = (_Float16*)(ws + WS_EMB16);
  _Float16* zf16  = (_Float16*)(ws + WS_ZF16);
  float*    enorm = (float*)(ws + WS_ENORM);
  unsigned* idxws = (unsigned*)(ws + WS_IDX);
  float*    parts = (float*)(ws + WS_PART);

  k_prep_emb<<<KCB / 8, 256, 0, stream>>>(emb, emb16, enorm);
  k_transpose_z<<<dim3(32, 8, NB), 256, 0, stream>>>(z, zf16);
  k_argmin<<<NVEC / 128, 256, 0, stream>>>(zf16, emb16, enorm, idxws,
                                           out + IDX1_OFF, out + IDX2_OFF);
  k_gather<<<1024, 256, 0, stream>>>(z, emb, idxws, out, parts);
  k_loss<<<1, 256, 0, stream>>>(parts, out + LOSS_OFF);
}
